// EMAQuantizer_31808527794305
// MI455X (gfx1250) — matrix operand order repaired and checked by reading, not compiled, not run on hardware
//
#include <hip/hip_runtime.h>

typedef __attribute__((ext_vector_type(16))) __bf16          v16bf;
typedef __attribute__((ext_vector_type(8)))  float           v8f;
typedef __attribute__((ext_vector_type(16))) unsigned short  u16x16;
typedef __attribute__((ext_vector_type(8)))  unsigned short  u16x8;

#define EMB_K   512          // embedding dim
#define NCODES  1024         // codebook entries
#define NROWS   32768        // B*H*W
#define QOUT    16777216     // B*C*H*W output floats

// round-to-nearest-even f32 -> bf16 bits
__device__ __forceinline__ unsigned int f2bf_bits(float x) {
  unsigned int u = __float_as_uint(x);
  return (u + 0x7FFFu + ((u >> 16) & 1u)) >> 16;
}

__device__ __forceinline__ v8f zero_v8f() {
  v8f v;
#pragma unroll
  for (int e = 0; e < 8; ++e) v[e] = 0.0f;
  return v;
}

// ---------------- Kernel 1: codebook -> bf16 hi/lo + 0.5*||e||^2 ----------------
__global__ __launch_bounds__(128) void vq_prep(const float* __restrict__ emb,
                                               unsigned short* __restrict__ ehi,
                                               unsigned short* __restrict__ elo,
                                               float* __restrict__ enorm) {
  const int k = blockIdx.x;                     // code id
  const float* row = emb + (size_t)k * EMB_K;
  float ss = 0.0f;
  for (int i = threadIdx.x; i < EMB_K; i += 128) {
    float x = row[i];
    unsigned int hb = f2bf_bits(x);
    float hf = __uint_as_float(hb << 16);
    unsigned int lb = f2bf_bits(x - hf);
    ehi[(size_t)k * EMB_K + i] = (unsigned short)hb;
    elo[(size_t)k * EMB_K + i] = (unsigned short)lb;
    ss += x * x;
  }
#pragma unroll
  for (int m = 16; m >= 1; m >>= 1) ss += __shfl_xor(ss, m, 32);
  __shared__ float red[4];
  if ((threadIdx.x & 31) == 0) red[threadIdx.x >> 5] = ss;
  __syncthreads();
  if (threadIdx.x == 0) enorm[k] = 0.5f * (red[0] + red[1] + red[2] + red[3]);
}

// ---------------- Kernel 2: distances via WMMA + per-row argmin ----------------
__global__ __launch_bounds__(128) void vq_dist(const float* __restrict__ z,
                                               const unsigned short* __restrict__ ehi,
                                               const unsigned short* __restrict__ elo,
                                               const float* __restrict__ enorm,
                                               int* __restrict__ idxOut,
                                               float* __restrict__ idxOutF) {
  constexpr int RS = 520;                       // padded LDS row stride (u16)
  __shared__ __align__(16) unsigned short sZhi[64 * RS];
  __shared__ __align__(16) unsigned short sZlo[64 * RS];

  const int t = threadIdx.x;
  // ---- stage: transpose NCHW z rows for this block into LDS as bf16 hi/lo ----
  {
    const int w = t & 31, g = t >> 5;           // lane reads consecutive W -> coalesced
    const int bh0 = blockIdx.x * 2;             // two (b,h) pairs per block = 64 rows
    for (int task = g; task < 2 * EMB_K; task += 4) {
      int p = task & 1, c = task >> 1;
      int bh = bh0 + p;
      int b = bh >> 5, h = bh & 31;
      float x = z[((size_t)b * EMB_K + c) * 1024 + (size_t)h * 32 + w];
      unsigned int hb = f2bf_bits(x);
      float hf = __uint_as_float(hb << 16);
      unsigned int lb = f2bf_bits(x - hf);
      int r = p * 32 + w;                       // local row 0..63
      sZhi[r * RS + c] = (unsigned short)hb;
      sZlo[r * RS + c] = (unsigned short)lb;
    }
  }
  __syncthreads();

  const int lane = t & 31;
  const int wave = t >> 5;
  const int l16  = lane & 15;
  const int rowBase = blockIdx.x * 64 + wave * 16;
  const unsigned short* zh = &sZhi[(wave * 16 + l16) * RS];
  const unsigned short* zl = &sZlo[(wave * 16 + l16) * RS];
  const int kOffA = (lane < 16) ? 0 : 16;       // A: contiguous k = 16h + s (matches B)
  const int kOffB = (lane < 16) ? 0 : 16;       // B: contiguous k = 16h + s (matches A)

  float rm[8];
  int   ri[8];
#pragma unroll
  for (int j = 0; j < 8; ++j) { rm[j] = 3.0e38f; ri[j] = 0; }

  constexpr int CT = 4;                         // code tiles per chunk
  for (int chunk = 0; chunk < NCODES / (16 * CT); ++chunk) {
    v8f acc[CT];
#pragma unroll
    for (int ct = 0; ct < CT; ++ct) acc[ct] = zero_v8f();

#pragma unroll 4
    for (int kt = 0; kt < EMB_K / 32; ++kt) {
      const int kb = kt * 32;
      u16x8 h0 = *(const u16x8*)(zh + kb + kOffA);
      u16x8 h1 = *(const u16x8*)(zh + kb + 8 + kOffA);
      u16x8 l0 = *(const u16x8*)(zl + kb + kOffA);
      u16x8 l1 = *(const u16x8*)(zl + kb + 8 + kOffA);
      u16x16 hv, lv;
#pragma unroll
      for (int e = 0; e < 8; ++e) {
        hv[e] = h0[e]; hv[e + 8] = h1[e];
        lv[e] = l0[e]; lv[e + 8] = l1[e];
      }
      v16bf Ahi = __builtin_bit_cast(v16bf, hv);
      v16bf Alo = __builtin_bit_cast(v16bf, lv);
#pragma unroll
      for (int ct = 0; ct < CT; ++ct) {
        const int code = (chunk * CT + ct) * 16 + l16;
        const size_t boff = (size_t)code * EMB_K + kb + kOffB;
        v16bf Bhi = __builtin_bit_cast(v16bf, *(const u16x16*)(ehi + boff));
        v16bf Blo = __builtin_bit_cast(v16bf, *(const u16x16*)(elo + boff));
        acc[ct] = __builtin_amdgcn_wmma_f32_16x16x32_bf16(false, Ahi, false, Bhi,
                                                          (short)0, acc[ct], false, false);
        acc[ct] = __builtin_amdgcn_wmma_f32_16x16x32_bf16(false, Ahi, false, Blo,
                                                          (short)0, acc[ct], false, false);
        acc[ct] = __builtin_amdgcn_wmma_f32_16x16x32_bf16(false, Alo, false, Bhi,
                                                          (short)0, acc[ct], false, false);
      }
    }

    // score = 0.5*||e||^2 - z.e  (monotonic in true distance); running argmin
#pragma unroll
    for (int ct = 0; ct < CT; ++ct) {
      const int code = (chunk * CT + ct) * 16 + l16;
      const float en = enorm[code];
#pragma unroll
      for (int j = 0; j < 8; ++j) {
        float s = en - acc[ct][j];
        if (s < rm[j]) { rm[j] = s; ri[j] = code; }   // strict < => first-min tie-break
      }
    }
  }

  // cross-lane min-reduce within 16-lane groups (rows live on same VGPR across lanes)
#pragma unroll
  for (int j = 0; j < 8; ++j) {
    float v = rm[j]; int i = ri[j];
#pragma unroll
    for (int m = 1; m < 16; m <<= 1) {
      float ov = __shfl_xor(v, m, 32);
      int   oi = __shfl_xor(i, m, 32);
      if (ov < v || (ov == v && oi < i)) { v = ov; i = oi; }
    }
    rm[j] = v; ri[j] = i;
  }
  if (l16 == 0) {
    const int rb = rowBase + (lane >> 4) * 8;   // lane0 -> rows 0..7, lane16 -> rows 8..15
#pragma unroll
    for (int j = 0; j < 8; ++j) {
      idxOut[rb + j]  = ri[j];
      idxOutF[rb + j] = (float)ri[j];
    }
  }
}

// ---------------- Kernel 3: gather embedding[idx] -> NCHW output ----------------
__global__ __launch_bounds__(256) void vq_gather(const float* __restrict__ emb,
                                                 const int* __restrict__ idx,
                                                 float* __restrict__ out) {
  const int bh = blockIdx.x;                    // b*32 + h
  const int b = bh >> 5, h = bh & 31;
  const int w  = threadIdx.x & 31;
  const int cj = threadIdx.x >> 5;              // 0..7
  const int n = (bh << 5) | w;
  const int code = idx[n];
  const float* erow = emb + (size_t)code * EMB_K;
  float* obase = out + (size_t)b * EMB_K * 1024 + (size_t)h * 32 + w;
  for (int c = cj; c < EMB_K; c += 8)
    obase[(size_t)c * 1024] = erow[c];          // 32 lanes -> 128B contiguous store
  if (blockIdx.x == 0 && threadIdx.x == 0)
    out[QOUT] = 0.0f;                           // tuple's scalar 0.0
}

extern "C" void kernel_launch(void* const* d_in, const int* in_sizes, int n_in,
                              void* d_out, int out_size, void* d_ws, size_t ws_size,
                              hipStream_t stream) {
  (void)in_sizes; (void)n_in; (void)out_size; (void)ws_size;
  const float* z   = (const float*)d_in[0];     // [32,512,32,32]
  const float* emb = (const float*)d_in[1];     // [1024,512]
  float* out = (float*)d_out;                   // quantized | 0.0 | indices(as f32)

  unsigned short* ehi  = (unsigned short*)d_ws;
  unsigned short* elo  = ehi + (size_t)NCODES * EMB_K;
  float*          enrm = (float*)(elo + (size_t)NCODES * EMB_K);
  int*            idx  = (int*)(enrm + NCODES);  // ~2.23 MB workspace total

  vq_prep  <<<NCODES,       128, 0, stream>>>(emb, ehi, elo, enrm);
  vq_dist  <<<NROWS / 64,   128, 0, stream>>>(z, ehi, elo, enrm, idx, out + QOUT + 1);
  vq_gather<<<32 * 32,      256, 0, stream>>>(emb, idx, out);
}
